// StackedLSTM_77232101916677
// MI455X (gfx1250) — matrix operand order repaired and checked by reading, not compiled, not run on hardware
//
#include <hip/hip_runtime.h>
#include <math.h>

// ---------------------------------------------------------------------------
// Stacked LSTM (3 layers, B=32, T=512, D=H=1024) for gfx1250 / MI455X.
// bf16 WMMA (v_wmma_f32_16x16x32_bf16) with f32 accumulate.
//  - weights converted once per call to bf16 (50MB -> resident in 192MB L2)
//  - per (t,layer) step kernel: 16 blocks x 8 waves.
//    wave pair = (x-projection wave, h-projection wave) K-split; each wave
//    covers M=32 (both batch tiles) so every weight fragment feeds 2 WMMAs
//    -> minimum 16.8MB weight traffic per step (the recurrence bottleneck).
//  - partial-sum reduction through LDS, fused branch-free LSTM cell epilogue.
// ---------------------------------------------------------------------------

#define Bsz   32
#define Tlen  512
#define Din   1024
#define Hd    1024
#define Lc    3          // effective layers (num_layers - 1)
#define G4H   4096       // 4*H
#define LDSP  1032       // padded LDS row pitch (bf16 elems) -> conflict-free

typedef __attribute__((ext_vector_type(16))) __bf16 v16bf;
typedef __attribute__((ext_vector_type(8)))  float  v8f;

union Frag {
    v16bf v;
    unsigned int   u[8];
    unsigned short s[16];
};

static __device__ __forceinline__ unsigned short f32_to_bf16(float f) {
    union { float f; unsigned int u; } x; x.f = f;
    unsigned int r = x.u + 0x7FFFu + ((x.u >> 16) & 1u);   // round-to-nearest-even
    return (unsigned short)(r >> 16);
}

// Branch-free transcendentals: v_exp_f32 + v_rcp_f32, saturate correctly.
static __device__ __forceinline__ float sigmoid_(float z) {
    return __builtin_amdgcn_rcpf(1.0f + __expf(-z));
}
static __device__ __forceinline__ float tanh_(float z) {
    return 1.0f - 2.0f * __builtin_amdgcn_rcpf(1.0f + __expf(2.0f * z));
}

// --------------------------- prep kernels ----------------------------------

__global__ __launch_bounds__(256) void cvt_weights(
    const float* __restrict__ Wih, const float* __restrict__ Whh,
    unsigned short* __restrict__ WihB, unsigned short* __restrict__ WhhB, long n)
{
    long stride = (long)gridDim.x * blockDim.x;
    for (long i = (long)blockIdx.x * blockDim.x + threadIdx.x; i < n; i += stride) {
        WihB[i] = f32_to_bf16(Wih[i]);
        WhhB[i] = f32_to_bf16(Whh[i]);
    }
}

__global__ __launch_bounds__(256) void sum_bias(
    const float* __restrict__ bih, const float* __restrict__ bhh,
    float* __restrict__ bias, int n)
{
    int stride = gridDim.x * blockDim.x;
    for (int i = blockIdx.x * blockDim.x + threadIdx.x; i < n; i += stride)
        bias[i] = bih[i] + bhh[i];
}

__global__ __launch_bounds__(256) void zero_state(float* __restrict__ p, long n)
{
    long stride = (long)gridDim.x * blockDim.x;
    for (long i = (long)blockIdx.x * blockDim.x + threadIdx.x; i < n; i += stride)
        p[i] = 0.0f;
}

__global__ __launch_bounds__(256) void final_state(
    const float* __restrict__ hfin, const float* __restrict__ cfin,
    float* __restrict__ out, long n)  // out points at d_out + B*T*H
{
    long stride = (long)gridDim.x * blockDim.x;
    for (long i = (long)blockIdx.x * blockDim.x + threadIdx.x; i < n; i += stride) {
        out[i]     = hfin[i];
        out[n + i] = cfin[i];
    }
}

// --------------------------- fused LSTM step -------------------------------
// grid = 16 blocks x 256 threads = 128 waves = 64 wave-pairs.
// pair -> j_tile in [0,64); within a pair: wave 0 computes x@Wih^T partials,
// wave 1 computes h@Whh^T partials, each over M=32 x (4 gate) 16x16 tiles.
// Partials combine through LDS; wave 0 applies the LSTM cell in-register.

__global__ __launch_bounds__(256) void lstm_step(
    const float* __restrict__ x, long ldx,          // input rows [32, 1024]
    const float* __restrict__ hprev,                // [32, H]
    const unsigned short* __restrict__ Wih,         // bf16 [4H, 1024]
    const unsigned short* __restrict__ Whh,         // bf16 [4H, H]
    const float* __restrict__ bias,                 // [4H]
    float* __restrict__ cstate,                     // [32, H] read+write
    float* __restrict__ hnext,                      // [32, H]
    float* __restrict__ outp)                       // d_out + t*H, or nullptr
{
    extern __shared__ unsigned short smem[];
    unsigned short* xs = smem;                  // [32][LDSP] bf16
    unsigned short* hs = smem + 32 * LDSP;      // [32][LDSP] bf16

    const int tid = threadIdx.x;

    // Stage x and h_prev into LDS as bf16 (coalesced global reads).
    for (int i = tid; i < 32 * 1024; i += 256) {
        int r = i >> 10, c = i & 1023;
        xs[r * LDSP + c] = f32_to_bf16(x[(long)r * ldx + c]);
        hs[r * LDSP + c] = f32_to_bf16(hprev[r * Hd + c]);
    }
    __syncthreads();

    const int lane   = tid & 31;
    const int wl     = tid >> 5;                 // wave in block: 0..7
    const int srcw   = wl & 1;                   // 0: x/Wih   1: h/Whh
    const int pair   = wl >> 1;                  // 0..3
    const int j_tile = (blockIdx.x << 2) + pair; // 0..63
    const int nn     = lane & 15;
    const int hh     = lane >> 4;
    const int col    = j_tile * 16 + nn;         // column within H

    // 8 accumulator tiles: [m_tile 0..1][gate 0..3]; x-wave seeds the bias.
    v8f acc[8];
#pragma unroll
    for (int g = 0; g < 4; ++g) {
        float bv = srcw ? 0.0f : bias[g * Hd + col];
        v8f t = { bv, bv, bv, bv, bv, bv, bv, bv };
        acc[g]     = t;
        acc[4 + g] = t;
    }

    const unsigned short* S = srcw ? hs  : xs;   // LDS activations (this wave's K-half)
    const unsigned short* W = srcw ? Whh : Wih;  // bf16 weights
    const int a0row = (lane & 15) * LDSP;        // batch rows 0..15
    const int a1row = (16 + (lane & 15)) * LDSP; // batch rows 16..31

    for (int kt = 0; kt < 32; ++kt) {
        const int k0 = kt << 5;
        // A fragments: 16-bit A 16x32 ISA layout; 2 bf16 per b32 DS load.
        Frag a0, a1;
#pragma unroll
        for (int p = 0; p < 8; ++p) {
            int kk = k0 + ((p < 4) ? (p << 1) : (16 + ((p - 4) << 1))) + (hh << 3);
            a0.u[p] = *(const unsigned int*)(&S[a0row + kk]);
            a1.u[p] = *(const unsigned int*)(&S[a1row + kk]);
        }
        // Each weight fragment (two 16B chunks at k 8h and 16+8h) feeds both batch tiles.
#pragma unroll
        for (int g = 0; g < 4; ++g) {
            Frag b;
            const uint4* wp =
                (const uint4*)(&W[(long)(g * Hd + col) * 1024 + k0 + (hh << 3)]);
            ((uint4*)&b)[0] = wp[0];
            ((uint4*)&b)[1] = wp[2];
            acc[g] = __builtin_amdgcn_wmma_f32_16x16x32_bf16(
                false, a0.v, false, b.v, (short)0, acc[g], false, false);
            acc[4 + g] = __builtin_amdgcn_wmma_f32_16x16x32_bf16(
                false, a1.v, false, b.v, (short)0, acc[4 + g], false, false);
        }
    }

    // Pair reduction through LDS (staging buffers are dead now).
    __syncthreads();
    float* red = (float*)smem;   // [pair][tile 0..7][r 0..7][lane] : 32KB
    if (srcw == 1) {
#pragma unroll
        for (int t8 = 0; t8 < 8; ++t8)
#pragma unroll
            for (int r = 0; r < 8; ++r)
                red[(((pair << 3) + t8) * 8 + r) * 32 + lane] = acc[t8][r];
    }
    __syncthreads();

    if (srcw == 0) {
#pragma unroll
        for (int t8 = 0; t8 < 8; ++t8)
#pragma unroll
            for (int r = 0; r < 8; ++r)
                acc[t8][r] += red[(((pair << 3) + t8) * 8 + r) * 32 + lane];

        // Fused cell: c' = sig(f)*c + sig(i)*tanh(g); h' = sig(o)*tanh(c').
#pragma unroll
        for (int m = 0; m < 2; ++m) {
#pragma unroll
            for (int r = 0; r < 8; ++r) {
                int  brow = m * 16 + r + (hh << 3);     // batch row
                long idx  = (long)brow * Hd + col;
                float iv = sigmoid_(acc[m * 4 + 0][r]);
                float fv = sigmoid_(acc[m * 4 + 1][r]);
                float gv = tanh_(acc[m * 4 + 2][r]);
                float ov = sigmoid_(acc[m * 4 + 3][r]);
                float cn = fv * cstate[idx] + iv * gv;
                cstate[idx] = cn;
                float hn = ov * tanh_(cn);
                hnext[idx] = hn;
                if (outp) outp[(long)brow * ((long)Tlen * Hd) + col] = hn;
            }
        }
    }
}

// --------------------------- host-side launch ------------------------------

extern "C" void kernel_launch(void* const* d_in, const int* in_sizes, int n_in,
                              void* d_out, int out_size, void* d_ws, size_t ws_size,
                              hipStream_t stream)
{
    (void)in_sizes; (void)n_in; (void)out_size; (void)ws_size;

    const float* in_seq = (const float*)d_in[0];   // [B,T,D]
    const float* Wih    = (const float*)d_in[1];   // [L,4H,D]
    const float* Whh    = (const float*)d_in[2];   // [L,4H,H]
    const float* bih    = (const float*)d_in[3];   // [L,4H]
    const float* bhh    = (const float*)d_in[4];   // [L,4H]
    float* out = (float*)d_out;

    // Workspace layout (all offsets 32B aligned).
    char* ws = (char*)d_ws;
    const long WN = (long)Lc * G4H * 1024;                 // 12,582,912 elems
    unsigned short* WihB = (unsigned short*)ws;            // bf16 weights
    unsigned short* WhhB = WihB + WN;
    float* bias = (float*)(ws + 2 * WN * sizeof(unsigned short));
    float* hbuf = bias + (long)Lc * G4H;                   // [2][L][B][H] ping-pong
    float* cbuf = hbuf + 2L * Lc * Bsz * Hd;               // [L][B][H]

    // 1) weights -> bf16, bias sum, zero state (re-done every call: stateless).
    cvt_weights<<<2048, 256, 0, stream>>>(Wih, Whh, WihB, WhhB, WN);
    sum_bias   <<<48,   256, 0, stream>>>(bih, bhh, bias, Lc * G4H);
    zero_state <<<288,  256, 0, stream>>>(hbuf, (2L + 1L) * Lc * Bsz * Hd);

    const size_t SMEM = 2ul * 32 * LDSP * sizeof(unsigned short);  // 132,096 B

    // 2) recurrence: stream-ordered (t, layer) chain of fused WMMA step kernels.
    for (int t = 0; t < Tlen; ++t) {
        int p = t & 1;
        for (int l = 0; l < Lc; ++l) {
            const float* x; long ldx;
            if (l == 0) { x = in_seq + (long)t * Din; ldx = (long)Tlen * Din; }
            else        { x = hbuf + ((long)(1 - p) * Lc + (l - 1)) * Bsz * Hd; ldx = Hd; }
            const float* hp = hbuf + ((long)p       * Lc + l) * Bsz * Hd;
            float*       hn = hbuf + ((long)(1 - p) * Lc + l) * Bsz * Hd;
            float*       cs = cbuf + (long)l * Bsz * Hd;
            const unsigned short* wi = WihB + (long)l * G4H * 1024;
            const unsigned short* wh = WhhB + (long)l * G4H * 1024;
            const float* bs = bias + (long)l * G4H;
            float* op = (l == Lc - 1) ? (out + (long)t * Hd) : nullptr;
            lstm_step<<<16, 256, SMEM, stream>>>(x, ldx, hp, wi, wh, bs, cs, hn, op);
        }
    }

    // 3) final states -> tail of d_out (h after t=511 lives in parity buffer 0).
    final_state<<<96, 256, 0, stream>>>(hbuf, cbuf,
                                        out + (long)Bsz * Tlen * Hd,
                                        (long)Lc * Bsz * Hd);
}
